// HardNegativeInfoNCELossWithLabels_64828236365828
// MI455X (gfx1250) — matrix operand order repaired and checked by reading, not compiled, not run on hardware
//
#include <hip/hip_runtime.h>

// ---------------------------------------------------------------------------
// HardNegativeInfoNCELossWithLabels for MI455X (gfx1250, wave32, WMMA).
// sim = normalize(A) @ normalize(S)^T / 0.05 computed via bf16x3-split WMMA
// (hi*hi + hi*lo + lo*hi) with fused flash-style weighted logsumexp.
// Round 4: keep 32x64 per-wave patch (2 row-tiles x 4 col-tiles, 24 WMMA per
// K-chunk) but pin the K loop to no-unroll so the allocator stays under 256
// VGPRs -> no scratch spills, no s_set_vgpr_msb churn in the hot loop.
// ---------------------------------------------------------------------------

typedef __attribute__((ext_vector_type(8)))  float  v8f;
typedef __attribute__((ext_vector_type(8)))  __bf16 v8bf;
typedef __attribute__((ext_vector_type(16))) __bf16 v16bf;

#define EMB_D        512
#define INV_TEMP     20.0f                 // 1 / 0.05
#define LOG_NEG_W    0.6931471805599453f   // log(2.0)
#define MARGIN       0.2f
#define NEG_BIG      3.0e38f
#define RTILES       2                     // row tiles per wave patch
#define CTILES       4                     // column tiles per wave patch
#define BLK_ROWS     (16 * RTILES)         // 32 rows per workgroup

// ---------------------------------------------------------------------------
// Kernel 1: L2-normalize each row and split fp32 -> bf16 hi + bf16 lo.
// One block (128 threads = 4 waves) per row.
// ---------------------------------------------------------------------------
__global__ __launch_bounds__(128)
void normalize_split_kernel(const float* __restrict__ in,
                            __bf16* __restrict__ hi,
                            __bf16* __restrict__ lo,
                            int D) {
  const int row = blockIdx.x;
  const float* r = in + (size_t)row * D;

  float ss = 0.f;
  for (int i = threadIdx.x; i < D; i += blockDim.x) {
    float v = r[i];
    ss = fmaf(v, v, ss);
  }
  for (int off = 16; off; off >>= 1) ss += __shfl_xor(ss, off, 32);
  __shared__ float red[4];
  const int wid = threadIdx.x >> 5;
  if ((threadIdx.x & 31) == 0) red[wid] = ss;
  __syncthreads();
  float tot = red[0] + red[1] + red[2] + red[3];
  const float inv = rsqrtf(tot);

  for (int i = threadIdx.x; i < D; i += blockDim.x) {
    float x = r[i] * inv;
    __bf16 h = (__bf16)x;
    hi[(size_t)row * D + i] = h;
    lo[(size_t)row * D + i] = (__bf16)(x - (float)h);
  }
}

// ---------------------------------------------------------------------------
// Kernel 2: per-column log-weights (explicit negatives get weight 2 -> log 2).
// ---------------------------------------------------------------------------
__global__ void logw_kernel(const int* __restrict__ labels,
                            float* __restrict__ logw, int N) {
  int i = blockIdx.x * blockDim.x + threadIdx.x;
  if (i < N) logw[i] = (labels[i] == 0) ? LOG_NEG_W : 0.0f;
}

// ---------------------------------------------------------------------------
// Kernel 3: fused GEMM (bf16x3 WMMA) + weighted online logsumexp per row.
// grid.x = N/32 row-blocks, 256 threads = 8 waves. 32-row A block staged in
// LDS (64 KB); each wave owns a 32x64 output patch (2 row-tiles x 4 col-tiles)
// and strides over the N/64 column strips. EXEC is uniform at every WMMA.
// ---------------------------------------------------------------------------
__global__ __launch_bounds__(256)
void sim_lse_kernel(const __bf16* __restrict__ Ahi, const __bf16* __restrict__ Alo,
                    const __bf16* __restrict__ Shi, const __bf16* __restrict__ Slo,
                    const float* __restrict__ logw,
                    float* __restrict__ lse, float* __restrict__ diag,
                    int N) {
  const int D = EMB_D;
  __shared__ __align__(32) __bf16 aHi[BLK_ROWS * EMB_D];   // 32 KB
  __shared__ __align__(32) __bf16 aLo[BLK_ROWS * EMB_D];   // 32 KB
  __shared__ float redM[8 * BLK_ROWS];
  __shared__ float redL[8 * BLK_ROWS];

  const int grow0 = blockIdx.x * BLK_ROWS;

  // Stage A row-block (hi and lo) into LDS, row-major [32][512].
  for (int idx = threadIdx.x; idx < BLK_ROWS * D; idx += blockDim.x) {
    int rr = idx >> 9;          // / 512
    int kk = idx & (D - 1);
    size_t g = (size_t)(grow0 + rr) * D + kk;
    aHi[idx] = Ahi[g];
    aLo[idx] = Alo[g];
  }
  __syncthreads();

  const int wave = threadIdx.x >> 5;
  const int lane = threadIdx.x & 31;
  const int half = lane >> 4;     // 0: lanes 0-15, 1: lanes 16-31
  const int l15  = lane & 15;

  // running (max, sum-exp): row = 16*r + v + 8*half, replicated across a half
  float rm[RTILES][8], rl[RTILES][8];
#pragma unroll
  for (int r = 0; r < RTILES; ++r)
#pragma unroll
    for (int j = 0; j < 8; ++j) { rm[r][j] = -NEG_BIG; rl[r][j] = 0.f; }

  // A fragment bases in LDS: row-tile r covers rows 16r..16r+15;
  // in contiguous k order (k = 16*half + s, as B) this lane reads row (16r + l15), K offset 16*half.
  const __bf16* aHiRow[RTILES];
  const __bf16* aLoRow[RTILES];
#pragma unroll
  for (int r = 0; r < RTILES; ++r) {
    aHiRow[r] = aHi + (16 * r + l15) * D + 16 * half;
    aLoRow[r] = aLo + (16 * r + l15) * D + 16 * half;
  }

  const int numStrips = N >> 6;           // N / 64
  for (int g = wave; g < numStrips; g += 8) {
    const int c0 = g << 6;                // first column of this 32x64 patch

    v8f acc[RTILES][CTILES];
#pragma unroll
    for (int r = 0; r < RTILES; ++r)
#pragma unroll
      for (int t = 0; t < CTILES; ++t) acc[r][t] = (v8f){};

    // B fragment base: column c0+l15 (+16 per tile), K offset 16*half.
    const __bf16* sHiBase = Shi + (size_t)(c0 + l15) * D + 16 * half;
    const __bf16* sLoBase = Slo + (size_t)(c0 + l15) * D + 16 * half;

    // No unroll: one iteration already exposes 24 WMMAs + 24 loads of ILP;
    // deeper pipelining overflows the 256-VGPR window and spills (round 3).
#pragma unroll 1
    for (int kc = 0; kc < EMB_D / 32; ++kc) {
      union { v16bf v; v8bf h[2]; } ah[RTILES], al[RTILES];
#pragma unroll
      for (int r = 0; r < RTILES; ++r) {
        ah[r].h[0] = *(const v8bf*)(aHiRow[r] + 32 * kc);      // K chunk 0
        ah[r].h[1] = *(const v8bf*)(aHiRow[r] + 32 * kc + 8);  // K chunk 1
        al[r].h[0] = *(const v8bf*)(aLoRow[r] + 32 * kc);
        al[r].h[1] = *(const v8bf*)(aLoRow[r] + 32 * kc + 8);
      }

#pragma unroll
      for (int t = 0; t < CTILES; ++t) {
        const size_t co = (size_t)t * 16 * D + 32 * kc;  // +16 columns per tile
        v16bf bh = *(const v16bf*)(sHiBase + co);
        v16bf bl = *(const v16bf*)(sLoBase + co);
#pragma unroll
        for (int r = 0; r < RTILES; ++r) {
          // sim += hi*hi + hi*lo + lo*hi  (lo*lo ~ 2^-32, dropped)
          acc[r][t] = __builtin_amdgcn_wmma_f32_16x16x32_bf16(
              false, ah[r].v, false, bh, (short)0, acc[r][t], false, false);
          acc[r][t] = __builtin_amdgcn_wmma_f32_16x16x32_bf16(
              false, ah[r].v, false, bl, (short)0, acc[r][t], false, false);
          acc[r][t] = __builtin_amdgcn_wmma_f32_16x16x32_bf16(
              false, al[r].v, false, bh, (short)0, acc[r][t], false, false);
        }
      }
    }

    // Epilogue: scale, add column weight (not on diagonal), online softmax.
#pragma unroll
    for (int t = 0; t < CTILES; ++t) {
      const int gcol = c0 + 16 * t + l15;
      const float lw = logw[gcol];
#pragma unroll
      for (int r = 0; r < RTILES; ++r) {
#pragma unroll
        for (int v = 0; v < 8; ++v) {
          const int M    = 16 * r + v + 8 * half;
          const int grow = grow0 + M;
          const float val = acc[r][t][v] * INV_TEMP;
          float wv;
          if (gcol == grow) {       // diagonal logit: weight 1, record sim
            diag[grow] = val;
            wv = val;
          } else {
            wv = val + lw;
          }
          // reduce across the 16 lanes of this half (row M lives there)
          float tmax = wv;
          tmax = fmaxf(tmax, __shfl_xor(tmax, 1, 32));
          tmax = fmaxf(tmax, __shfl_xor(tmax, 2, 32));
          tmax = fmaxf(tmax, __shfl_xor(tmax, 4, 32));
          tmax = fmaxf(tmax, __shfl_xor(tmax, 8, 32));
          float tsum = __expf(wv - tmax);
          tsum += __shfl_xor(tsum, 1, 32);
          tsum += __shfl_xor(tsum, 2, 32);
          tsum += __shfl_xor(tsum, 4, 32);
          tsum += __shfl_xor(tsum, 8, 32);
          const float nm = fmaxf(rm[r][v], tmax);
          rl[r][v] = rl[r][v] * __expf(rm[r][v] - nm) + tsum * __expf(tmax - nm);
          rm[r][v] = nm;
        }
      }
    }
  }

  // Combine the 8 waves' partial (max, sum) per row via LDS.
  if (l15 == 0) {
#pragma unroll
    for (int r = 0; r < RTILES; ++r)
#pragma unroll
      for (int v = 0; v < 8; ++v) {
        const int M = 16 * r + v + 8 * half;
        redM[wave * BLK_ROWS + M] = rm[r][v];
        redL[wave * BLK_ROWS + M] = rl[r][v];
      }
  }
  __syncthreads();
  if (threadIdx.x < BLK_ROWS) {
    float Mx = -NEG_BIG;
    for (int w = 0; w < 8; ++w)
      Mx = fmaxf(Mx, redM[w * BLK_ROWS + threadIdx.x]);
    float L = 0.f;
    for (int w = 0; w < 8; ++w)
      L += redL[w * BLK_ROWS + threadIdx.x] *
           __expf(redM[w * BLK_ROWS + threadIdx.x] - Mx);
    lse[grow0 + threadIdx.x] = Mx + __logf(L);
  }
}

// ---------------------------------------------------------------------------
// Kernel 4: final scalar reductions -> loss.
// ---------------------------------------------------------------------------
__global__ __launch_bounds__(1024)
void finalize_kernel(const int* __restrict__ labels,
                     const float* __restrict__ diag,
                     const float* __restrict__ lse,
                     float* __restrict__ out, int N) {
  __shared__ float sLoss[32], sDiag[32], sNeg[32];
  __shared__ int   sPos[32],  sNegC[32];

  float loss = 0.f, dpos = 0.f, negmax = -NEG_BIG;
  int cpos = 0, cneg = 0;
  for (int i = threadIdx.x; i < N; i += blockDim.x) {
    const int   lb = labels[i];
    const float d  = diag[i];
    if (lb == 1) { loss += lse[i] - d; dpos += d; ++cpos; }
    else         { negmax = fmaxf(negmax, d); ++cneg; }
  }
  for (int off = 16; off; off >>= 1) {
    loss   += __shfl_xor(loss, off, 32);
    dpos   += __shfl_xor(dpos, off, 32);
    negmax  = fmaxf(negmax, __shfl_xor(negmax, off, 32));
    cpos   += __shfl_xor(cpos, off, 32);
    cneg   += __shfl_xor(cneg, off, 32);
  }
  const int wid = threadIdx.x >> 5;
  if ((threadIdx.x & 31) == 0) {
    sLoss[wid] = loss; sDiag[wid] = dpos; sNeg[wid] = negmax;
    sPos[wid] = cpos;  sNegC[wid] = cneg;
  }
  __syncthreads();
  if (threadIdx.x == 0) {
    float L = 0.f, Dp = 0.f, Nm = -NEG_BIG;
    int Cp = 0, Cn = 0;
    const int nw = blockDim.x >> 5;
    for (int w = 0; w < nw; ++w) {
      L += sLoss[w]; Dp += sDiag[w]; Nm = fmaxf(Nm, sNeg[w]);
      Cp += sPos[w]; Cn += sNegC[w];
    }
    const float fp = (float)Cp;
    const float infonce  = L / fp;
    const float mean_pos = Dp / fp;
    float pen = fmaxf(Nm - mean_pos + MARGIN, 0.f);
    if (Cn == 0) pen = 0.f;
    out[0] = infonce + pen;   // HARD_NEG_WEIGHT == 1.0
  }
}

// ---------------------------------------------------------------------------
// Host-side launch. Workspace layout (256B-aligned):
//   Ahi, Alo, Shi, Slo : N*D bf16 each  (4 * 8.39 MB)
//   logw, diag, lse    : N f32 each
// ---------------------------------------------------------------------------
extern "C" void kernel_launch(void* const* d_in, const int* in_sizes, int n_in,
                              void* d_out, int out_size, void* d_ws, size_t ws_size,
                              hipStream_t stream) {
  const float* a      = (const float*)d_in[0];
  const float* s      = (const float*)d_in[1];
  const int*   labels = (const int*)d_in[2];   // per-harness: integer -> const int*

  const int N = in_sizes[2];   // 8192
  const int D = EMB_D;         // 512 (in_sizes[0] / N)

  char* ws = (char*)d_ws;
  size_t off = 0;
  auto take = [&](size_t bytes) -> char* {
    char* p = ws + off;
    off = (off + bytes + 255) & ~(size_t)255;
    return p;
  };
  const size_t bfBytes = (size_t)N * D * sizeof(__bf16);
  __bf16* Ahi = (__bf16*)take(bfBytes);
  __bf16* Alo = (__bf16*)take(bfBytes);
  __bf16* Shi = (__bf16*)take(bfBytes);
  __bf16* Slo = (__bf16*)take(bfBytes);
  float*  logw = (float*)take((size_t)N * sizeof(float));
  float*  diag = (float*)take((size_t)N * sizeof(float));
  float*  lse  = (float*)take((size_t)N * sizeof(float));

  normalize_split_kernel<<<N, 128, 0, stream>>>(a, Ahi, Alo, D);
  normalize_split_kernel<<<N, 128, 0, stream>>>(s, Shi, Slo, D);
  logw_kernel<<<(N + 255) / 256, 256, 0, stream>>>(labels, logw, N);
  sim_lse_kernel<<<N / BLK_ROWS, 256, 0, stream>>>(Ahi, Alo, Shi, Slo,
                                                   logw, lse, diag, N);
  finalize_kernel<<<1, 1024, 0, stream>>>(labels, diag, lse, (float*)d_out, N);
}
